// GRAM_59983513255966
// MI455X (gfx1250) — compile-verified, matrix operand order checked by reading, not run on hardware
//
#include <hip/hip_runtime.h>
#include <hip/hip_bf16.h>
#include <math.h>

#define B_ 8
#define C_ 512
#define E_ 512
#define H_ 48
#define W_ 48
#define N_ (H_*W_)       // 2304
#define P_ 128
#define D_ 128           // E/4
#define KNUM_ 230        // int(N * 0.1)
#define BNF 0.9999950000374997f   // 1/sqrt(1+1e-5): eval BN, running stats 0/1
#define SAF_ 68          // LDS fp32 row stride (272B: 16B-aligned, bank-break)

typedef _Float16 half_t;
typedef __attribute__((ext_vector_type(16))) _Float16 v16h;
typedef __attribute__((ext_vector_type(8)))  _Float16 v8h;
typedef __attribute__((ext_vector_type(8)))  float    v8f;

// ---- CDNA5 async global->LDS copies (ASYNCcnt) --------------------------
// LDS destination is a per-lane byte offset (generic shared ptr low 32 bits:
// ISA 10.2 aperture).
__device__ __forceinline__ void async_ld_b128(unsigned ldsOff, const void* g) {
  asm volatile("global_load_async_to_lds_b128 %0, %1, off"
               :: "v"(ldsOff), "v"(g) : "memory");
}
__device__ __forceinline__ void async_ld_b32(unsigned ldsOff, const void* g) {
  asm volatile("global_load_async_to_lds_b32 %0, %1, off"
               :: "v"(ldsOff), "v"(g) : "memory");
}
#define WAIT_ASYNC(n) asm volatile("s_wait_asynccnt " #n ::: "memory")

// ---------------------------------------------------------------------------
// Tiled WMMA GEMM: Y[b][m][n] = post( sum_k A[m,k] * Xsel[k,n] )
//  - 128 threads (4 waves), 32x32 macro-tile, K-tile 64.
//  - Double-buffered async staging of fp32 tiles into LDS (TDM-style overlap):
//    issue tile t+1, s_wait_asynccnt <=OPS for tile t, compute 2 WMMAs.
//  - ATRANS: A[m,k] = A[k*lda + m] (attention q^T k, lda=N); staged via
//    per-lane b32 scatter (free transpose). B always scattered to [n][k].
//  - SPLIT:  k < ksplit -> X1, else X2 (concat without materializing)
//  - MODE:   0 none, 1 +bias, 2 relu(bn), 3 bn   (gb=gamma/bias, bt=beta)
// Fragment layouts per CDNA5 ISA 7.12.2.
// ---------------------------------------------------------------------------
template<int ATRANS, int MODE, int SPLIT>
__global__ __launch_bounds__(128) void gemm32_wmma(
    const float* __restrict__ A, const float* __restrict__ X1,
    const float* __restrict__ X2, float* __restrict__ Y,
    int K, int Nn, int lda,
    long aBS, long x1BS, long x2BS, long yBS,
    int ksplit, const float* __restrict__ gb, const float* __restrict__ bt,
    float postScale)
{
  __shared__ float fA[2][32 * SAF_];
  __shared__ float fB[2][32 * SAF_];

  const int tid  = threadIdx.x;
  const int wave = tid >> 5;
  const int lane = tid & 31;
  const int hlf  = lane >> 4;
  const int r    = lane & 15;
  const int wm   = (wave >> 1) << 4;     // wave quadrant row offset (0/16)
  const int wn   = (wave & 1) << 4;      // wave quadrant col offset (0/16)
  const int n0   = blockIdx.x << 5;
  const int m0   = blockIdx.y << 5;
  const int b    = blockIdx.z;
  const float* Ab  = A  + (long)b * aBS;
  const float* X1b = X1 + (long)b * x1BS;
  const float* X2b = X2 + (long)b * x2BS;

  // issue all async copies for one 64-K tile into LDS buffer bi
  auto stage = [&](int bi, int k0) {
    float* dA = &fA[bi][0];
    float* dB = &fB[bi][0];
    if (ATRANS) {
      // A global [k][m] (coalesced along m): b32 scatter -> dA[m][k]
#pragma unroll
      for (int i = 0; i < 16; ++i) {
        const int kr = wave + (i << 2);
        async_ld_b32((unsigned)(size_t)&dA[lane * SAF_ + kr],
                     Ab + (long)(k0 + kr) * lda + m0 + lane);
      }
    } else {
      // A global [m][k]: direct b128 rows -> dA[m][k]
      const int row = tid >> 2;                 // m 0..31
      const int sk  = (tid & 3) << 4;           // k 0/16/32/48
      const float* src = Ab + (long)(m0 + row) * lda + k0 + sk;
      float* dst = &dA[row * SAF_ + sk];
#pragma unroll
      for (int i = 0; i < 4; ++i)
        async_ld_b128((unsigned)(size_t)(dst + (i << 2)), src + (i << 2));
    }
    // B global [k][n] (coalesced along n): b32 scatter -> dB[n][k]
#pragma unroll
    for (int i = 0; i < 16; ++i) {
      const int kr = wave + (i << 2);
      const int kg = k0 + kr;
      const float* Xp;
      if (SPLIT) Xp = (kg < ksplit) ? (X1b + (long)kg * Nn)
                                    : (X2b + (long)(kg - ksplit) * Nn);
      else       Xp = X1b + (long)kg * Nn;
      async_ld_b32((unsigned)(size_t)&dB[lane * SAF_ + kr], Xp + n0 + lane);
    }
  };

  const int T = K >> 6;                  // 64-K tiles
  v8f acc = {};
  stage(0, 0);
  for (int t = 0; t < T; ++t) {
    if (t + 1 < T) {
      stage((t + 1) & 1, (t + 1) << 6);  // prefetch next tile (other buffer)
      if (ATRANS) WAIT_ASYNC(32);        // current tile's 32 ops retired
      else        WAIT_ASYNC(20);        // current tile's 20 ops retired
    } else {
      WAIT_ASYNC(0);
    }
    __syncthreads();

    const float* pa = &fA[t & 1][(wm + r) * SAF_];
    const float* pb = &fB[t & 1][(wn + r) * SAF_];
#pragma unroll
    for (int kb = 0; kb < 64; kb += 32) {
      v16h af, bf;
#pragma unroll
      for (int e = 0; e < 16; ++e) {
        // element e -> K offset (VGPR e/2 slot e%2): (e&8)*2 + hlf*8 + (e&7)
        const int kk = kb + ((e & 8) << 1) + (hlf << 3) + (e & 7);
        af[e] = (half_t)pa[kk];
        bf[e] = (half_t)pb[kk];
      }
      acc = __builtin_amdgcn_wmma_f32_16x16x32_f16(false, af, false, bf,
                                                   (short)0, acc, false, false);
    }
    __syncthreads();   // safe to overwrite this buffer two tiles later
  }

#pragma unroll
  for (int j = 0; j < 8; ++j) {
    const int m = m0 + wm + (hlf << 3) + j;
    float y = acc[j] * postScale;
    if (MODE == 1)      y += gb[m];
    else if (MODE == 2) y = fmaxf(fmaf(y, gb[m] * BNF, bt[m]), 0.f);
    else if (MODE == 3) y = fmaf(y, gb[m] * BNF, bt[m]);
    Y[(long)b * yBS + (long)m * Nn + (n0 + wn + r)] = y;
  }
}

// ---------------------------------------------------------------------------
// Depthwise 3x3 + eval-BN + ReLU,  in/out: [B,E,H,W], w: [E,1,3,3]
// ---------------------------------------------------------------------------
__global__ __launch_bounds__(256) void dwconv_bn_relu(
    const float* __restrict__ in, const float* __restrict__ w,
    const float* __restrict__ g,  const float* __restrict__ bb,
    float* __restrict__ out)
{
  const int idx = blockIdx.x * blockDim.x + threadIdx.x;
  if (idx >= B_ * E_ * N_) return;
  const int wv = idx % W_;
  const int hv = (idx / W_) % H_;
  const int c  = (idx / N_) % E_;
  const int b  = idx / (E_ * N_);
  const float* ip = in + ((long)b * E_ + c) * N_;
  const float* wp = w + c * 9;
  float s = 0.f;
#pragma unroll
  for (int i = 0; i < 3; ++i) {
    const int hh = hv + i - 1;
    if (hh < 0 || hh >= H_) continue;
#pragma unroll
    for (int j = 0; j < 3; ++j) {
      const int ww = wv + j - 1;
      if (ww < 0 || ww >= W_) continue;
      s = fmaf(ip[hh * W_ + ww], wp[i * 3 + j], s);
    }
  }
  out[idx] = fmaxf(fmaf(s, g[c] * BNF, bb[c]), 0.f);
}

// ---------------------------------------------------------------------------
// ContextBlock attention pooling mask: logits + softmax over N, per batch
// ---------------------------------------------------------------------------
__global__ __launch_bounds__(256) void mask_softmax(
    const float* __restrict__ x, const float* __restrict__ cmw,
    const float* __restrict__ cmb, float* __restrict__ mask)
{
  const int b = blockIdx.x, tid = threadIdx.x;
  __shared__ float red[256];
  const float* xb = x + (long)b * C_ * N_;
  float lmax = -1e30f;
  for (int n = tid; n < N_; n += 256) {
    float s = cmb[0];
    for (int c = 0; c < C_; ++c) s = fmaf(xb[(long)c * N_ + n], cmw[c], s);
    mask[b * N_ + n] = s;
    lmax = fmaxf(lmax, s);
  }
  red[tid] = lmax; __syncthreads();
  for (int off = 128; off > 0; off >>= 1) { if (tid < off) red[tid] = fmaxf(red[tid], red[tid + off]); __syncthreads(); }
  const float mx = red[0]; __syncthreads();
  float ls = 0.f;
  for (int n = tid; n < N_; n += 256) { float e = __expf(mask[b * N_ + n] - mx); mask[b * N_ + n] = e; ls += e; }
  red[tid] = ls; __syncthreads();
  for (int off = 128; off > 0; off >>= 1) { if (tid < off) red[tid] += red[tid + off]; __syncthreads(); }
  const float inv = 1.f / red[0]; __syncthreads();
  for (int n = tid; n < N_; n += 256) mask[b * N_ + n] *= inv;
}

// out[b,c] = sum_n mat[b,c,n] * vec[b,n]   (one wave per (b,c))
__global__ __launch_bounds__(256) void pool_dot(
    const float* __restrict__ mat, const float* __restrict__ vec,
    float* __restrict__ out, int R)
{
  const int gw = (blockIdx.x * blockDim.x + threadIdx.x) >> 5;
  const int lane = threadIdx.x & 31;
  if (gw >= B_ * R) return;
  const int b = gw / R, c = gw % R;
  const float* mp = mat + ((long)b * R + c) * N_;
  const float* vp = vec + (long)b * N_;
  float s = 0.f;
  for (int n = lane; n < N_; n += 32) s = fmaf(mp[n], vp[n], s);
  for (int off = 16; off > 0; off >>= 1) s += __shfl_down(s, off);
  if (lane == 0) out[gw] = s;
}

// ---------------------------------------------------------------------------
// Both ContextBlock channel MLPs (conv1x1 -> LN -> ReLU -> conv1x1), per batch
// pass 0 -> channel_add; pass 1 -> sigmoid(channel_mul)
// ---------------------------------------------------------------------------
__global__ __launch_bounds__(128) void chan_mlp2(
    const float* __restrict__ ctx,
    const float* __restrict__ aw1, const float* __restrict__ ab1,
    const float* __restrict__ alg, const float* __restrict__ alb,
    const float* __restrict__ aw2, const float* __restrict__ ab2,
    const float* __restrict__ mw1, const float* __restrict__ mb1,
    const float* __restrict__ mlg, const float* __restrict__ mlb,
    const float* __restrict__ mw2, const float* __restrict__ mb2,
    float* __restrict__ addv, float* __restrict__ mulv)
{
  const int b = blockIdx.x, tid = threadIdx.x;   // 128 threads
  __shared__ float sc[C_];
  __shared__ float h[P_];
  __shared__ float red[P_];
  for (int c = tid; c < C_; c += P_) sc[c] = ctx[b * C_ + c];
  __syncthreads();
  for (int pass = 0; pass < 2; ++pass) {
    const float* w1 = pass ? mw1 : aw1;  const float* b1 = pass ? mb1 : ab1;
    const float* lg = pass ? mlg : alg;  const float* lb = pass ? mlb : alb;
    const float* w2 = pass ? mw2 : aw2;  const float* b2 = pass ? mb2 : ab2;
    float s = b1[tid];
    for (int c = 0; c < C_; ++c) s = fmaf(sc[c], w1[tid * C_ + c], s);
    red[tid] = s; __syncthreads();
    for (int off = 64; off > 0; off >>= 1) { if (tid < off) red[tid] += red[tid + off]; __syncthreads(); }
    const float mu = red[0] / (float)P_; __syncthreads();
    const float d = s - mu;
    red[tid] = d * d; __syncthreads();
    for (int off = 64; off > 0; off >>= 1) { if (tid < off) red[tid] += red[tid + off]; __syncthreads(); }
    const float var = red[0] / (float)P_; __syncthreads();
    h[tid] = fmaxf(fmaf(d * rsqrtf(var + 1e-5f), lg[tid], lb[tid]), 0.f);
    __syncthreads();
    for (int c = tid; c < C_; c += P_) {
      float o = b2[c];
      for (int p2 = 0; p2 < P_; ++p2) o = fmaf(h[p2], w2[c * P_ + p2], o);
      if (pass == 0) addv[b * C_ + c] = o;
      else           mulv[b * C_ + c] = 1.f / (1.f + __expf(-o));
    }
    __syncthreads();
  }
}

// glob = x * mul[b,c] + add[b,c]
__global__ __launch_bounds__(256) void glob_ew(
    const float* __restrict__ x, const float* __restrict__ mulv,
    const float* __restrict__ addv, float* __restrict__ out)
{
  const int idx = blockIdx.x * blockDim.x + threadIdx.x;
  if (idx >= B_ * C_ * N_) return;
  const int c = (idx / N_) % C_;
  const int b = idx / (C_ * N_);
  out[idx] = fmaf(x[idx], mulv[b * C_ + c], addv[b * C_ + c]);
}

// sem (2ch) and bdl (2ch) heads from gx
__global__ __launch_bounds__(256) void heads_k(
    const float* __restrict__ gx,
    const float* __restrict__ shw, const float* __restrict__ shb,
    const float* __restrict__ bhw, const float* __restrict__ bhb,
    float* __restrict__ sem, float* __restrict__ bdl)
{
  const int idx = blockIdx.x * blockDim.x + threadIdx.x;
  if (idx >= B_ * N_) return;
  const int b = idx / N_, n = idx % N_;
  const float* g = gx + (long)b * E_ * N_ + n;
  float s0 = shb[0], s1 = shb[1], d0 = bhb[0], d1 = bhb[1];
  for (int c = 0; c < E_; ++c) {
    const float v = g[(long)c * N_];
    s0 = fmaf(v, shw[c], s0);       s1 = fmaf(v, shw[E_ + c], s1);
    d0 = fmaf(v, bhw[c], d0);       d1 = fmaf(v, bhw[E_ + c], d1);
  }
  sem[((long)b * 2 + 0) * N_ + n] = s0;  sem[((long)b * 2 + 1) * N_ + n] = s1;
  bdl[((long)b * 2 + 0) * N_ + n] = d0;  bdl[((long)b * 2 + 1) * N_ + n] = d1;
}

// wx = v * (1 - sigmoid(v))
__global__ __launch_bounds__(256) void wx_ew(const float* __restrict__ in, float* __restrict__ out)
{
  const int idx = blockIdx.x * blockDim.x + threadIdx.x;
  if (idx >= B_ * 2 * N_) return;
  const float v = in[idx];
  out[idx] = v * (1.f - 1.f / (1.f + __expf(-v)));
}

// 2ch -> 2ch conv3x3 + bias + eval-BN + ReLU
__global__ __launch_bounds__(256) void conv3x3_2(
    const float* __restrict__ in, const float* __restrict__ w,
    const float* __restrict__ cb, const float* __restrict__ g,
    const float* __restrict__ bb, float* __restrict__ out)
{
  const int idx = blockIdx.x * blockDim.x + threadIdx.x;
  if (idx >= B_ * 2 * N_) return;
  const int wv = idx % W_;
  const int hv = (idx / W_) % H_;
  const int o  = (idx / N_) % 2;
  const int b  = idx / (2 * N_);
  float s = cb[o];
#pragma unroll
  for (int ci = 0; ci < 2; ++ci) {
    const float* ip = in + ((long)b * 2 + ci) * N_;
    const float* wp = w + (o * 2 + ci) * 9;
#pragma unroll
    for (int i = 0; i < 3; ++i) {
      const int hh = hv + i - 1;
      if (hh < 0 || hh >= H_) continue;
#pragma unroll
      for (int j = 0; j < 3; ++j) {
        const int ww = wv + j - 1;
        if (ww < 0 || ww >= W_) continue;
        s = fmaf(ip[hh * W_ + ww], wp[i * 3 + j], s);
      }
    }
  }
  out[idx] = fmaxf(fmaf(s, g[o] * BNF, bb[o]), 0.f);
}

// region = (1-st)*(1-bt), st/bt from self-cosine > NC (faithful to source)
__global__ __launch_bounds__(256) void region_k(
    const float* __restrict__ sem, const float* __restrict__ r2,
    float* __restrict__ region)
{
  const int idx = blockIdx.x * blockDim.x + threadIdx.x;
  if (idx >= B_ * N_) return;
  const int b = idx / N_, n = idx % N_;
  float a0 = sem[((long)b * 2 + 0) * N_ + n], a1 = sem[((long)b * 2 + 1) * N_ + n];
  float n2 = a0 * a0 + a1 * a1;
  float nn = fmaxf(sqrtf(n2), 1e-8f);
  const float st = (n2 / (nn * nn) > 2.0f) ? 1.f : 0.f;
  a0 = r2[((long)b * 2 + 0) * N_ + n];  a1 = r2[((long)b * 2 + 1) * N_ + n];
  n2 = a0 * a0 + a1 * a1;
  nn = fmaxf(sqrtf(n2), 1e-8f);
  const float bt = (n2 / (nn * nn) > 2.0f) ? 1.f : 0.f;
  region[idx] = (1.f - st) * (1.f - bt);
}

// ---------------------------------------------------------------------------
// Per-row softmax (in-place -> probabilities) + exact top-k mean via 4-pass
// byte-radix select with LDS histogram.  attn: one [N_,N_] slab (one batch).
// ---------------------------------------------------------------------------
__global__ __launch_bounds__(256) void row_softmax_topk(
    float* __restrict__ attn, float* __restrict__ soft, int knum)
{
  const int m = blockIdx.x, tid = threadIdx.x;
  float* row = attn + (long)m * N_;
  __shared__ float red[256];
  __shared__ unsigned hist[256];
  __shared__ unsigned s_digit;
  __shared__ int s_rem;

  float lmax = -1e30f;
  for (int n = tid; n < N_; n += 256) lmax = fmaxf(lmax, row[n]);
  red[tid] = lmax; __syncthreads();
  for (int off = 128; off > 0; off >>= 1) { if (tid < off) red[tid] = fmaxf(red[tid], red[tid + off]); __syncthreads(); }
  const float mx = red[0]; __syncthreads();

  float ls = 0.f;
  for (int n = tid; n < N_; n += 256) { const float e = __expf(row[n] - mx); row[n] = e; ls += e; }
  red[tid] = ls; __syncthreads();
  for (int off = 128; off > 0; off >>= 1) { if (tid < off) red[tid] += red[tid + off]; __syncthreads(); }
  const float inv = 1.f / red[0]; __syncthreads();
  for (int n = tid; n < N_; n += 256) row[n] *= inv;
  __syncthreads();

  // radix select k-th largest (probs >= 0 so uint order == float order)
  unsigned prefix = 0;
  int remaining = knum;
  for (int pass = 0; pass < 4; ++pass) {
    const int shift = 24 - pass * 8;
    hist[tid] = 0u; __syncthreads();
    for (int n = tid; n < N_; n += 256) {
      const unsigned u = __float_as_uint(row[n]);
      const bool sel = (pass == 0) ? true : ((u >> (shift + 8)) == prefix);
      if (sel) atomicAdd(&hist[(u >> shift) & 255u], 1u);
    }
    __syncthreads();
    if (tid == 0) {
      int acc = 0, d = 255;
      for (; d > 0; --d) { const int hc = (int)hist[d]; if (acc + hc >= remaining) break; acc += hc; }
      s_digit = (unsigned)d;
      s_rem = remaining - acc;
    }
    __syncthreads();
    prefix = (prefix << 8) | s_digit;
    remaining = s_rem;
    __syncthreads();
  }
  // exact top-k sum: strictly-greater values + ties at threshold
  float sgt = 0.f;
  for (int n = tid; n < N_; n += 256) {
    if (__float_as_uint(row[n]) > prefix) sgt += row[n];
  }
  red[tid] = sgt; __syncthreads();
  for (int off = 128; off > 0; off >>= 1) { if (tid < off) red[tid] += red[tid + off]; __syncthreads(); }
  if (tid == 0)
    soft[m] = (red[0] + (float)remaining * __uint_as_float(prefix)) / (float)knum;
}

// colfac[n] = soft[n] * mean_m(prob[m,n])
__global__ __launch_bounds__(256) void col_factor(
    const float* __restrict__ attn, const float* __restrict__ soft,
    float* __restrict__ colfac)
{
  const int n = blockIdx.x * blockDim.x + threadIdx.x;
  if (n >= N_) return;
  float s = 0.f;
  for (int m = 0; m < N_; ++m) s += attn[(long)m * N_ + n];
  colfac[n] = soft[n] * s * (1.f / (float)N_);
}

// X2[b,c,n] = ctx2[b,c] * region[b,n]   (rank-1 second half of final concat)
__global__ __launch_bounds__(256) void x2_mat(
    const float* __restrict__ ctx2, const float* __restrict__ region,
    float* __restrict__ out)
{
  const int idx = blockIdx.x * blockDim.x + threadIdx.x;
  if (idx >= B_ * E_ * N_) return;
  const int n = idx % N_;
  const int c = (idx / N_) % E_;
  const int b = idx / (E_ * N_);
  out[idx] = ctx2[b * E_ + c] * region[b * N_ + n];
}

// ---------------------------------------------------------------------------
extern "C" void kernel_launch(void* const* d_in, const int* in_sizes, int n_in,
                              void* d_out, int out_size, void* d_ws, size_t ws_size,
                              hipStream_t stream) {
  (void)in_sizes; (void)n_in; (void)out_size; (void)ws_size;
  const float* x      = (const float*)d_in[0];
  const float* lb_w1  = (const float*)d_in[1];
  const float* lb_g1  = (const float*)d_in[2];
  const float* lb_b1  = (const float*)d_in[3];
  const float* lb_w2  = (const float*)d_in[4];
  const float* lb_g2  = (const float*)d_in[5];
  const float* lb_b2  = (const float*)d_in[6];
  const float* lb_w3  = (const float*)d_in[7];
  const float* lb_g3  = (const float*)d_in[8];
  const float* lb_b3  = (const float*)d_in[9];
  const float* cm_w   = (const float*)d_in[10];
  const float* cm_b   = (const float*)d_in[11];
  const float* ca_w1  = (const float*)d_in[12];
  const float* ca_b1  = (const float*)d_in[13];
  const float* ca_lg  = (const float*)d_in[14];
  const float* ca_lb  = (const float*)d_in[15];
  const float* ca_w2  = (const float*)d_in[16];
  const float* ca_b2  = (const float*)d_in[17];
  const float* cmu_w1 = (const float*)d_in[18];
  const float* cmu_b1 = (const float*)d_in[19];
  const float* cmu_lg = (const float*)d_in[20];
  const float* cmu_lb = (const float*)d_in[21];
  const float* cmu_w2 = (const float*)d_in[22];
  const float* cmu_b2 = (const float*)d_in[23];
  const float* fu_w   = (const float*)d_in[24];
  const float* fu_b   = (const float*)d_in[25];
  const float* sh_w   = (const float*)d_in[26];
  const float* sh_b   = (const float*)d_in[27];
  const float* bh_w   = (const float*)d_in[28];
  const float* bh_b   = (const float*)d_in[29];
  const float* ra_w1  = (const float*)d_in[30];
  const float* ra_b1  = (const float*)d_in[31];
  const float* ra_g1  = (const float*)d_in[32];
  const float* ra_bb1 = (const float*)d_in[33];
  const float* ra_w2  = (const float*)d_in[34];
  const float* ra_b2  = (const float*)d_in[35];
  const float* ra_g2  = (const float*)d_in[36];
  const float* ra_bb2 = (const float*)d_in[37];
  const float* q_w    = (const float*)d_in[38];
  const float* k_w    = (const float*)d_in[39];
  const float* v_w    = (const float*)d_in[40];
  const float* fin_w  = (const float*)d_in[41];

  // ---- workspace layout (floats) ----
  float* ws = (float*)d_ws;
  const size_t big = (size_t)B_ * E_ * N_;          // 9,437,184
  float* buf0 = ws;                                  // t / local / v
  float* buf1 = buf0 + big;                          // t2 / glob / attn slab / X2
  float* buf2 = buf1 + big;                          // gx
  float* qb   = buf2 + big;                          // [B,128,N]
  float* kb   = qb + (size_t)B_ * D_ * N_;
  float* p    = kb + (size_t)B_ * D_ * N_;
  float* maskb   = p; p += (size_t)B_ * N_;
  float* ctxb    = p; p += (size_t)B_ * C_;
  float* addv    = p; p += (size_t)B_ * C_;
  float* mulv    = p; p += (size_t)B_ * C_;
  float* semb    = p; p += (size_t)B_ * 2 * N_;
  float* bdlb    = p; p += (size_t)B_ * 2 * N_;
  float* wxb     = p; p += (size_t)B_ * 2 * N_;
  float* r1b     = p; p += (size_t)B_ * 2 * N_;
  float* r2b     = p; p += (size_t)B_ * 2 * N_;
  float* regionb = p; p += (size_t)B_ * N_;
  float* softb   = p; p += (size_t)B_ * N_;
  float* colfacb = p; p += (size_t)B_ * N_;
  float* ctx2b   = p; p += (size_t)B_ * E_;

  const dim3 gE(N_ / 32, E_ / 32, B_);     // 512-out GEMMs
  const dim3 gD(N_ / 32, D_ / 32, B_);     // 128-out GEMMs
  const dim3 gA(N_ / 32, N_ / 32, 1);      // attention logits (per batch)
  const long sEN = (long)E_ * N_, sDN = (long)D_ * N_, sCN = (long)C_ * N_;
  const int totE = B_ * E_ * N_, tot2 = B_ * 2 * N_, totBN = B_ * N_;

  // GLoR local branch: conv1x1 -> BN -> ReLU ; dw3x3 -> BN -> ReLU ; conv1x1 -> BN
  gemm32_wmma<0,2,0><<<gE, 128, 0, stream>>>(lb_w1, x, x, buf0, C_, N_, C_,
      0, sCN, sCN, sEN, C_, lb_g1, lb_b1, 1.f);
  dwconv_bn_relu<<<(totE + 255) / 256, 256, 0, stream>>>(buf0, lb_w2, lb_g2, lb_b2, buf1);
  gemm32_wmma<0,3,0><<<gE, 128, 0, stream>>>(lb_w3, buf1, buf1, buf0, E_, N_, E_,
      0, sEN, sEN, sEN, E_, lb_g3, lb_b3, 1.f);

  // ContextBlock
  mask_softmax<<<B_, 256, 0, stream>>>(x, cm_w, cm_b, maskb);
  pool_dot<<<(B_ * C_ * 32 + 255) / 256, 256, 0, stream>>>(x, maskb, ctxb, C_);
  chan_mlp2<<<B_, 128, 0, stream>>>(ctxb, ca_w1, ca_b1, ca_lg, ca_lb, ca_w2, ca_b2,
      cmu_w1, cmu_b1, cmu_lg, cmu_lb, cmu_w2, cmu_b2, addv, mulv);
  glob_ew<<<(totE + 255) / 256, 256, 0, stream>>>(x, mulv, addv, buf1);

  // fusion conv over concat[local, glob]  (split-K GEMM, +bias)
  gemm32_wmma<0,1,1><<<gE, 128, 0, stream>>>(fu_w, buf0, buf1, buf2, 2 * E_, N_, 2 * E_,
      0, sEN, sEN, sEN, E_, fu_b, fu_b, 1.f);

  // heads + ReverseAttention + BGAM region
  heads_k<<<(totBN + 255) / 256, 256, 0, stream>>>(buf2, sh_w, sh_b, bh_w, bh_b, semb, bdlb);
  wx_ew<<<(tot2 + 255) / 256, 256, 0, stream>>>(bdlb, wxb);
  conv3x3_2<<<(tot2 + 255) / 256, 256, 0, stream>>>(wxb, ra_w1, ra_b1, ra_g1, ra_bb1, r1b);
  conv3x3_2<<<(tot2 + 255) / 256, 256, 0, stream>>>(r1b, ra_w2, ra_b2, ra_g2, ra_bb2, r2b);
  region_k<<<(totBN + 255) / 256, 256, 0, stream>>>(semb, r2b, regionb);

  // AGCR projections
  gemm32_wmma<0,0,0><<<gD, 128, 0, stream>>>(q_w, buf2, buf2, qb, E_, N_, E_,
      0, sEN, sEN, sDN, E_, nullptr, nullptr, 1.f);
  gemm32_wmma<0,0,0><<<gD, 128, 0, stream>>>(k_w, buf2, buf2, kb, E_, N_, E_,
      0, sEN, sEN, sDN, E_, nullptr, nullptr, 1.f);
  gemm32_wmma<0,0,0><<<gE, 128, 0, stream>>>(v_w, buf2, buf2, buf0, E_, N_, E_,
      0, sEN, sEN, sEN, E_, nullptr, nullptr, 1.f);    // v -> buf0 (local dead)

  // attention: per-batch N x N slab in buf1 (glob dead)
  const float pscale = 0.08838834764831845f;   // 1/sqrt(128)
  for (int b = 0; b < B_; ++b) {
    gemm32_wmma<1,0,0><<<gA, 128, 0, stream>>>(qb + (size_t)b * D_ * N_,
        kb + (size_t)b * D_ * N_, kb + (size_t)b * D_ * N_, buf1,
        D_, N_, N_, 0, 0, 0, 0, D_, nullptr, nullptr, pscale);
    row_softmax_topk<<<N_, 256, 0, stream>>>(buf1, softb + (size_t)b * N_, KNUM_);
    col_factor<<<(N_ + 255) / 256, 256, 0, stream>>>(buf1,
        softb + (size_t)b * N_, colfacb + (size_t)b * N_);
  }

  // ctx2 = V @ colfac ; X2 rank-1 materialize ; final split-K GEMM -> d_out
  pool_dot<<<(B_ * E_ * 32 + 255) / 256, 256, 0, stream>>>(buf0, colfacb, ctx2b, E_);
  x2_mat<<<(totE + 255) / 256, 256, 0, stream>>>(ctx2b, regionb, buf1);
  gemm32_wmma<0,0,1><<<gE, 128, 0, stream>>>(fin_w, buf2, buf1, (float*)d_out,
      2 * E_, N_, 2 * E_, 0, sEN, sEN, sEN, E_, nullptr, nullptr, 1.f);
}
